// AssembledBlock_6167573037591
// MI455X (gfx1250) — compile-verified, matrix operand order checked by reading, not run on hardware
//
#include <hip/hip_runtime.h>

typedef __attribute__((ext_vector_type(16))) _Float16 v16h;
typedef __attribute__((ext_vector_type(8)))  _Float16 v8h;
typedef __attribute__((ext_vector_type(4)))  _Float16 v4h;
typedef __attribute__((ext_vector_type(8)))  float    v8f;

// ---------------------------------------------------------------------------
// Problem constants (from reference): B=16, C=OC=64, H=W=128, E=4, K=3, HID=16
// ---------------------------------------------------------------------------
#define BATCH 16
#define CH    64
#define HW    128
#define NEXP  4
#define NHID  16
#define KTAPS 9            // 3x3
#define KDIM  (KTAPS*CH)   // 576
#define KT_N  (KDIM/32)    // 18 k-tiles of 32

// ---------------------------------------------------------------------------
// 1) Global average pool: one block per (b,c), deterministic tree reduction
// ---------------------------------------------------------------------------
__global__ __launch_bounds__(256)
void pool_kernel(const float* __restrict__ x, float* __restrict__ pooled) {
  const int bc = blockIdx.x;                    // 0..1023
  const float4* p = (const float4*)(x + (size_t)bc * (HW * HW));
  float s = 0.f;
  for (int i = threadIdx.x; i < (HW * HW) / 4; i += 256) {
    float4 v = p[i];
    s += v.x + v.y + v.z + v.w;
  }
  __shared__ float red[256];
  red[threadIdx.x] = s;
  __syncthreads();
  for (int off = 128; off > 0; off >>= 1) {
    if ((int)threadIdx.x < off) red[threadIdx.x] += red[threadIdx.x + off];
    __syncthreads();
  }
  if (threadIdx.x == 0) pooled[bc] = red[0] * (1.f / (HW * HW));
}

// ---------------------------------------------------------------------------
// 2) Control MLP + softmax(T=30) + aggregated biases.  One block per sample.
//    coeff layout: [b][oc][e]   aggB layout: [layer][b][oc]
// ---------------------------------------------------------------------------
__global__ __launch_bounds__(256)
void control2_kernel(const float* __restrict__ pooled,
                     const float* __restrict__ cw1,  // (HID, C)
                     const float* __restrict__ cw2,  // (E*OC, HID)
                     const float* __restrict__ b1,   // (E, OC)
                     const float* __restrict__ b2,
                     const float* __restrict__ b3,
                     float* __restrict__ coeff,
                     float* __restrict__ aggB) {
  const int b = blockIdx.x;
  const int t = threadIdx.x;
  __shared__ float sp[CH], sh[NHID], sl[NEXP * CH];
  if (t < CH) sp[t] = pooled[b * CH + t];
  __syncthreads();
  if (t < NHID) {
    float s = 0.f;
    for (int c = 0; c < CH; ++c) s += sp[c] * cw1[t * CH + c];
    sh[t] = s > 0.f ? s : 0.f;
  }
  __syncthreads();
  {  // logits, j = oc*E + e  (reshape (B, OC, E))
    float s = 0.f;
    for (int h = 0; h < NHID; ++h) s += sh[h] * cw2[t * NHID + h];
    sl[t] = s;
  }
  __syncthreads();
  if (t < CH) {
    float l0 = sl[t * 4 + 0], l1 = sl[t * 4 + 1], l2 = sl[t * 4 + 2], l3 = sl[t * 4 + 3];
    float m = fmaxf(fmaxf(l0, l1), fmaxf(l2, l3));
    const float it = 1.f / 30.f;
    float e0 = __expf((l0 - m) * it), e1 = __expf((l1 - m) * it);
    float e2 = __expf((l2 - m) * it), e3 = __expf((l3 - m) * it);
    float inv = 1.f / (e0 + e1 + e2 + e3);
    float c0 = e0 * inv, c1 = e1 * inv, c2 = e2 * inv, c3 = e3 * inv;
    coeff[(b * CH + t) * 4 + 0] = c0;
    coeff[(b * CH + t) * 4 + 1] = c1;
    coeff[(b * CH + t) * 4 + 2] = c2;
    coeff[(b * CH + t) * 4 + 3] = c3;
    aggB[(0 * BATCH + b) * CH + t] = c0*b1[t] + c1*b1[CH+t] + c2*b1[2*CH+t] + c3*b1[3*CH+t];
    aggB[(1 * BATCH + b) * CH + t] = c0*b2[t] + c1*b2[CH+t] + c2*b2[2*CH+t] + c3*b2[3*CH+t];
    aggB[(2 * BATCH + b) * CH + t] = c0*b3[t] + c1*b3[CH+t] + c2*b3[2*CH+t] + c3*b3[3*CH+t];
  }
}

// ---------------------------------------------------------------------------
// 3) Expert-mix weights, emitted directly in WMMA A-fragment order (f16).
//    K ordering: k = tap*64 + c  (tap-major => each 32-wide k-tile = one tap,
//    32 contiguous channels).  Fragment: [l][b][oc_t][kt][lane][16 halves].
//    A 16x32 f16 layout: lanes 0-15 K-groups {0-7,16-23}; lanes 16-31 {8-15,24-31}.
// ---------------------------------------------------------------------------
__global__ __launch_bounds__(32)
void prep_w_kernel(const float* __restrict__ w1,  // (E, OC, C, 3, 3)
                   const float* __restrict__ w2,
                   const float* __restrict__ w3,
                   const float* __restrict__ coeff,
                   _Float16* __restrict__ fragW) {
  const int blk = blockIdx.x;                 // ((l*16+b)*4+oc_t)*18+kt
  const int kt   = blk % KT_N;
  const int oc_t = (blk / KT_N) & 3;
  const int b    = (blk / (KT_N * 4)) & (BATCH - 1);
  const int l    = blk / (KT_N * 4 * BATCH);
  const float* w = (l == 0) ? w1 : (l == 1) ? w2 : w3;

  const int lane = threadIdx.x;
  const int lh = lane >> 4;
  const int oc = oc_t * 16 + (lane & 15);

  float cf0 = coeff[(b * CH + oc) * 4 + 0];
  float cf1 = coeff[(b * CH + oc) * 4 + 1];
  float cf2 = coeff[(b * CH + oc) * 4 + 2];
  float cf3 = coeff[(b * CH + oc) * 4 + 3];

  v16h frag;
#pragma unroll
  for (int e = 0; e < 16; ++e) {
    int kap = (e < 8) ? (lh * 8 + e) : (16 + lh * 8 + (e - 8));
    int k = kt * 32 + kap;
    int c = k & 63;
    int tap = k >> 6;                         // 0..8  -> (ky*3+kx)
    size_t base = ((size_t)oc * CH + c) * KTAPS + tap;
    float v = cf0 * w[base]
            + cf1 * w[(size_t)CH * CH * KTAPS + base]
            + cf2 * w[2 * (size_t)CH * CH * KTAPS + base]
            + cf3 * w[3 * (size_t)CH * CH * KTAPS + base];
    frag[e] = (_Float16)v;
  }
  *((v16h*)fragW + ((size_t)blk * 32 + lane)) = frag;
}

// ---------------------------------------------------------------------------
// 4) fp32 -> f16 activation copy (input of layer 1)
// ---------------------------------------------------------------------------
__global__ __launch_bounds__(256)
void cvt_f32_f16_kernel(const float* __restrict__ in, _Float16* __restrict__ out, int n4) {
  int i = blockIdx.x * 256 + threadIdx.x;
  if (i < n4) {
    float4 v = ((const float4*)in)[i];
    v4h h = {(_Float16)v.x, (_Float16)v.y, (_Float16)v.z, (_Float16)v.w};
    *((v4h*)out + i) = h;
  }
}

// ---------------------------------------------------------------------------
// 5) 3x3 conv, pad 1, as implicit GEMM on v_wmma_f32_16x16x32_f16.
//    Block = (b, row y, 32-pixel segment).  8 waves = 4 oc-tiles x 2 px-tiles.
//    LDS patch: [ry(3)][col(34)][c(64)] f16 -> channel-contiguous, so each
//    B-fragment K-group (8 consecutive channels of one tap) is one b128 read.
// ---------------------------------------------------------------------------
__global__ __launch_bounds__(256)
void conv3x3_wmma_kernel(const _Float16* __restrict__ inH,
                         const _Float16* __restrict__ fragW,  // per-layer base
                         const float* __restrict__ aggB,      // per-layer [b][oc]
                         _Float16* __restrict__ outH,         // used if !write_f32
                         float* __restrict__ outF,            // used if write_f32
                         int write_f32) {
  __shared__ _Float16 sIn[3 * 34 * CH];  // 13056 B

  const int blk = blockIdx.x;
  const int xs = blk & 3;             // 4 segments of 32 px
  const int y  = (blk >> 2) & (HW - 1);
  const int b  = blk >> 9;
  const int t = threadIdx.x;

  // ---- stage input patch (3 rows x 34 cols x 64 ch) into LDS as f16 ----
  const int y0 = y - 1, xb = xs * 32 - 1;
  for (int j = t; j < 3 * 34 * CH; j += 256) {
    int col = j % 34;
    int rc  = j / 34;
    int c   = rc & 63;
    int ry  = rc >> 6;
    int yy = y0 + ry, xx = xb + col;
    _Float16 v = (_Float16)0.f;
    if ((unsigned)yy < (unsigned)HW && (unsigned)xx < (unsigned)HW)
      v = inH[(((size_t)(b * CH + c) * HW + yy) << 7) + xx];
    sIn[(ry * 34 + col) * CH + c] = v;
  }
  __syncthreads();

  const int wid  = t >> 5, lane = t & 31;
  const int oc_t = wid & 3, px_t = wid >> 2;
  const int n    = lane & 15, lh = lane >> 4;
  const int g0   = lh * 8;

  const v16h* Abase = (const v16h*)fragW + ((size_t)(b * 4 + oc_t) * (KT_N * 32) + lane);

  v8f acc = {0.f, 0.f, 0.f, 0.f, 0.f, 0.f, 0.f, 0.f};

#pragma unroll
  for (int kt = 0; kt < KT_N; ++kt) {
    if (kt + 1 < KT_N) __builtin_prefetch(Abase + (size_t)(kt + 1) * 32, 0, 0);
    v16h a = Abase[(size_t)kt * 32];

    const int tap = kt >> 1;             // two k-tiles per tap
    const int c0  = (kt & 1) << 5;       // channel half 0..31 / 32..63
    const int ky  = tap / 3;
    const int kx  = tap - ky * 3;
    const int col = px_t * 16 + n + kx;
    const int pos = (ky * 34 + col) * CH;

    v8h lo = *(const v8h*)&sIn[pos + c0 + g0];        // K-group {0-7 | 8-15}
    v8h hi = *(const v8h*)&sIn[pos + c0 + 16 + g0];   // K-group {16-23 | 24-31}
    v16h bb;
#pragma unroll
    for (int i = 0; i < 8; ++i) { bb[i] = lo[i]; bb[8 + i] = hi[i]; }

    acc = __builtin_amdgcn_wmma_f32_16x16x32_f16(
        /*neg_a=*/false, a, /*neg_b=*/false, bb,
        /*c_mod=*/(short)0, acc, /*reuse_a=*/false, /*reuse_b=*/false);
  }

  // ---- epilogue: bias + store (C/D layout: VGPR r -> M=r+8*lh, N=lane&15) --
  const int x = xs * 32 + px_t * 16 + n;
#pragma unroll
  for (int r = 0; r < 8; ++r) {
    int oc = oc_t * 16 + r + (lh ? 8 : 0);
    float v = acc[r] + aggB[b * CH + oc];
    size_t oidx = (((size_t)(b * CH + oc) * HW + y) << 7) + x;
    if (write_f32) outF[oidx] = v;
    else           outH[oidx] = (_Float16)v;
  }
}

// ---------------------------------------------------------------------------
// Launch
// ---------------------------------------------------------------------------
extern "C" void kernel_launch(void* const* d_in, const int* in_sizes, int n_in,
                              void* d_out, int out_size, void* d_ws, size_t ws_size,
                              hipStream_t stream) {
  const float* x   = (const float*)d_in[0];
  const float* cw1 = (const float*)d_in[1];
  const float* cw2 = (const float*)d_in[2];
  const float* w1  = (const float*)d_in[3];
  const float* w2  = (const float*)d_in[4];
  const float* w3  = (const float*)d_in[5];
  const float* b1  = (const float*)d_in[6];
  const float* b2  = (const float*)d_in[7];
  const float* b3  = (const float*)d_in[8];
  float* out = (float*)d_out;

  char* ws = (char*)d_ws;
  float*    pooled = (float*)(ws + 0);                 //  4 KB
  float*    coeff  = (float*)(ws + 4096);              // 16 KB
  float*    aggB   = (float*)(ws + 20480);             // 12 KB
  _Float16* fragW  = (_Float16*)(ws + 32768);          // 3 * 1.125 MB
  _Float16* xh_a   = (_Float16*)(ws + (size_t)(4u  << 20));  // 32 MB
  _Float16* xh_b   = (_Float16*)(ws + (size_t)(36u << 20));  // 32 MB

  const size_t FR   = (size_t)BATCH * 4 * KT_N * 512;  // halves per layer
  const int    BIAS = BATCH * CH;                      // floats per layer
  const int    NPIX = BATCH * CH * HW * HW;            // 16,777,216

  pool_kernel<<<BATCH * CH, 256, 0, stream>>>(x, pooled);
  control2_kernel<<<BATCH, 256, 0, stream>>>(pooled, cw1, cw2, b1, b2, b3, coeff, aggB);
  prep_w_kernel<<<3 * BATCH * 4 * KT_N, 32, 0, stream>>>(w1, w2, w3, coeff, fragW);
  cvt_f32_f16_kernel<<<(NPIX / 4 + 255) / 256, 256, 0, stream>>>(x, xh_a, NPIX / 4);

  const int CONV_GRID = BATCH * HW * 4;  // 8192 blocks
  conv3x3_wmma_kernel<<<CONV_GRID, 256, 0, stream>>>(xh_a, fragW + 0 * FR, aggB + 0 * BIAS, xh_b, nullptr, 0);
  conv3x3_wmma_kernel<<<CONV_GRID, 256, 0, stream>>>(xh_b, fragW + 1 * FR, aggB + 1 * BIAS, xh_a, nullptr, 0);
  conv3x3_wmma_kernel<<<CONV_GRID, 256, 0, stream>>>(xh_a, fragW + 2 * FR, aggB + 2 * BIAS, nullptr, out, 1);
}
